// ContrastiveLongformer_28973849378853
// MI455X (gfx1250) — matrix operand order repaired and checked by reading, not compiled, not run on hardware
//
#include <hip/hip_runtime.h>
#include <hip/hip_bf16.h>

// ---------------- model constants (longformer-base-4096) ----------------
#define BB   2
#define SS   4096
#define DD   768
#define LL   12
#define HH   12
#define FF   3072
#define CW   256          // one-sided window
#define DH   64
#define NTOK (BB*SS)      // 8192 rows

// ---------------- types for WMMA ----------------
typedef __attribute__((ext_vector_type(16))) __bf16 bf16x16;
typedef __attribute__((ext_vector_type(8)))  float  f32x8;
typedef __attribute__((ext_vector_type(4)))  unsigned int u32x4;
typedef __attribute__((ext_vector_type(4)))  int i32x4;

union FragU { bf16x16 v; u32x4 q[2]; };

__device__ __forceinline__ f32x8 wmma_bf16(FragU a, FragU b, f32x8 c) {
  // D = A(16x32 bf16) x B(32x16 bf16) + C(16x16 f32)
  return __builtin_amdgcn_wmma_f32_16x16x32_bf16(
      false, a.v, false, b.v, (short)0, c, false, false);
}

__device__ __forceinline__ unsigned short f2bf_bits(float f) {
  unsigned int u = __float_as_uint(f);
  unsigned int r = ((u >> 16) & 1u) + 0x7fffu;   // round-to-nearest-even
  return (unsigned short)((u + r) >> 16);
}

// ---------------- async global->LDS (CDNA5) with sync fallback ----------------
// Builtin signature (from clang diagnostic): (int4 AS1*, int4 AS3*, imm int, imm int).
// AS casts must go through integers; flat->LDS keeps the low 32 bits (aperture mapping).
typedef __attribute__((address_space(1))) i32x4 gi32x4;
typedef __attribute__((address_space(3))) i32x4 li32x4;

#if defined(__gfx1250__) && __has_builtin(__builtin_amdgcn_global_load_async_to_lds_b128)
#define ASYNC_LDS 1
#define ASYNC_CP16(g, l)                                             \
  __builtin_amdgcn_global_load_async_to_lds_b128(                    \
      (gi32x4*)(unsigned long long)(g),                              \
      (li32x4*)(unsigned int)(unsigned long long)(l), 0, 0)
#else
#define ASYNC_CP16(g, l) (*(u32x4*)(l) = *(const u32x4*)(g))
#endif

__device__ __forceinline__ void wait_async0() {
#if defined(ASYNC_LDS)
#if __has_builtin(__builtin_amdgcn_s_wait_asynccnt)
  __builtin_amdgcn_s_wait_asynccnt(0);
#else
  asm volatile("s_wait_asynccnt 0x0" ::: "memory");
#endif
#endif
}

// ---------------- f32 -> bf16, transposing per layer: [L,K,N] -> [L,N,K] ----------------
__global__ void cvtT_kernel(const float* __restrict__ in,
                            unsigned short* __restrict__ outp,
                            int K, int N, long long total) {
  long long i  = (long long)blockIdx.x * blockDim.x + threadIdx.x;
  long long st = (long long)gridDim.x * blockDim.x;
  long long kn = (long long)K * N;
  for (; i < total; i += st) {
    long long l = i / kn, rem = i - l*kn;
    int n_ = (int)(rem / K), k_ = (int)(rem - (long long)n_*K);
    outp[i] = f2bf_bits(in[l*kn + (size_t)k_*N + n_]);
  }
}

// ---------------- embeddings (gather + add) ----------------
__global__ __launch_bounds__(256) void embed_kernel(
    const int* __restrict__ ids, const float* __restrict__ wemb,
    const float* __restrict__ pemb, const float* __restrict__ temb,
    float* __restrict__ outp) {
  int row = blockIdx.x;               // b*SS + s
  int s = row % SS;
  int id = ids[row];
  for (int c = threadIdx.x; c < DD; c += 256)
    outp[(size_t)row*DD + c] =
        wemb[(size_t)id*DD + c] + pemb[(size_t)(s + 2)*DD + c] + temb[c];
}

// ---------------- fused (x [+bias] [+residual]) LayerNorm -> f32 + bf16 ----------------
__global__ __launch_bounds__(256) void ln_kernel(
    const float* __restrict__ x, const float* __restrict__ bias,
    const float* __restrict__ res, const float* __restrict__ g,
    const float* __restrict__ bb, float* __restrict__ ho,
    unsigned short* __restrict__ hb) {
  __shared__ float red[256];
  int row = blockIdx.x, tid = threadIdx.x;
  float vv[3];
  for (int e = 0; e < 3; ++e) {
    int c = tid + e*256;
    float t = x[(size_t)row*DD + c];
    if (bias) t += bias[c];
    if (res)  t += res[(size_t)row*DD + c];
    vv[e] = t;
  }
  red[tid] = vv[0] + vv[1] + vv[2];
  __syncthreads();
  for (int o = 128; o > 0; o >>= 1) { if (tid < o) red[tid] += red[tid+o]; __syncthreads(); }
  float mean = red[0] * (1.f/DD);
  __syncthreads();
  float ss = 0.f;
  for (int e = 0; e < 3; ++e) { float d = vv[e] - mean; ss += d*d; }
  red[tid] = ss;
  __syncthreads();
  for (int o = 128; o > 0; o >>= 1) { if (tid < o) red[tid] += red[tid+o]; __syncthreads(); }
  float rstd = rsqrtf(red[0]*(1.f/DD) + 1e-5f);
  for (int e = 0; e < 3; ++e) {
    int c = tid + e*256;
    float o = (vv[e] - mean)*rstd*g[c] + bb[c];
    ho[(size_t)row*DD + c] = o;
    hb[(size_t)row*DD + c] = f2bf_bits(o);
  }
}

// ---------------- bf16 WMMA GEMM: C = A(MxK) * B(KxN), B given pre-transposed (NxK) ----
// Epilogues (compile-time):
//   0: f32 out (no bias)          1: bf16 out + bias
//   2: bf16 out + bias + GELU     3: bf16 transposed out [B,H,DH,S] + bias (for V)
// 256 threads = 8 waves; block tile 128x128; K-step 32; double-buffered async LDS tiles.
template <int EPI>
__global__ __launch_bounds__(256) void gemm_bf16_kernel(
    const unsigned short* __restrict__ A,   // M x K bf16
    const unsigned short* __restrict__ BT,  // N x K bf16
    const float* __restrict__ bias,         // N (EPI != 0)
    float* __restrict__ Cf,                 // M x N f32 (EPI == 0)
    unsigned short* __restrict__ Cb,        // bf16 out
    int M, int N, int K) {
  __shared__ __align__(16) unsigned short smA[2][128*32];  // [m][k]
  __shared__ __align__(16) unsigned short smB[2][128*32];  // [n][k]
  int tid = threadIdx.x;
  int wave = tid >> 5, lane = tid & 31;
  int ml = lane & 15, hi = lane >> 4;
  int wr = wave >> 2, wc = wave & 3;
  int rbase = blockIdx.y * 128;
  int cbase = blockIdx.x * 128;

  auto stage = [&](int t, int buf) {
    int kk = t * 32;
    for (int it = 0; it < 2; ++it) {
      int u = tid + it*256;
      int row = u >> 2, kq = (u & 3)*8;
      ASYNC_CP16(A + (size_t)(rbase + row)*K + kk + kq, &smA[buf][row*32 + kq]);
    }
    for (int it = 0; it < 2; ++it) {
      int u = tid + it*256;
      int n_ = u >> 2, kq = (u & 3)*8;
      ASYNC_CP16(BT + (size_t)(cbase + n_)*K + kk + kq, &smB[buf][n_*32 + kq]);
    }
  };

  f32x8 acc[4][2] = {};
  int nk = K / 32;

  stage(0, 0);
  wait_async0();
  __syncthreads();

  for (int t = 0; t < nk; ++t) {
    int cur = t & 1;
    if (t + 1 < nk) stage(t + 1, cur ^ 1);   // overlap DMA with WMMA

    FragU bfr[2];
    for (int j = 0; j < 2; ++j) {
      int n = (wc*2 + j)*16 + ml;
      // B layout: lanes 0-15 K=0..15, lanes 16-31 K=16..31 (contiguous per lane)
      bfr[j].q[0] = *(const u32x4*)&smB[cur][n*32 + hi*16];
      bfr[j].q[1] = *(const u32x4*)&smB[cur][n*32 + hi*16 + 8];
    }
    for (int i = 0; i < 4; ++i) {
      FragU af;
      int m = (wr*4 + i)*16 + ml;
      // A layout: lanes 0-15 K=0..15, lanes 16-31 K=16..31 (contiguous per lane)
      af.q[0] = *(const u32x4*)&smA[cur][m*32 + hi*16];
      af.q[1] = *(const u32x4*)&smA[cur][m*32 + hi*16 + 8];
      for (int j = 0; j < 2; ++j)
        acc[i][j] = wmma_bf16(af, bfr[j], acc[i][j]);
    }
    wait_async0();
    __syncthreads();
  }

  // epilogue: C element (m = e + hi*8, n = ml) within each 16x16 tile
  for (int j = 0; j < 2; ++j) {
    int c = cbase + (wc*2 + j)*16 + ml;
    float bv = (EPI != 0) ? bias[c] : 0.f;
    for (int i = 0; i < 4; ++i) {
      int r0 = rbase + (wr*4 + i)*16 + hi*8;
      if (EPI == 0) {
        for (int e = 0; e < 8; ++e)
          Cf[(size_t)(r0 + e)*N + c] = acc[i][j][e];
      } else if (EPI == 3) {
        // V transposed: vt[b][h][d][s]; 8 consecutive rows -> one 16B store
        unsigned int w[4];
        for (int e2 = 0; e2 < 4; ++e2) {
          unsigned short lo = f2bf_bits(acc[i][j][2*e2]     + bv);
          unsigned short hs = f2bf_bits(acc[i][j][2*e2 + 1] + bv);
          w[e2] = (unsigned)lo | ((unsigned)hs << 16);
        }
        int b_ = r0 / SS, s0 = r0 % SS;
        int h_ = c / DH, d_ = c % DH;
        *(u32x4*)(Cb + (((size_t)(b_*HH + h_)*DH + d_)*SS + s0)) =
            (u32x4){w[0], w[1], w[2], w[3]};
      } else {
        for (int e = 0; e < 8; ++e) {
          float val = acc[i][j][e] + bv;
          if (EPI == 2) val = 0.5f * val * (1.f + erff(val * 0.70710678118f));
          Cb[(size_t)(r0 + e)*N + c] = f2bf_bits(val);
        }
      }
    }
  }
}

// ---------------- fused sliding-window attention ----------------
// One block = 64 query rows of one (b, head, 256-chunk). 256 threads = 8 waves.
// scores(64x768,K=64) -> mask/exp -> probs(LDS bf16) -> P*V(64x64,K=768)
// V is pre-transposed globally: vt[b][h][d][s], so V^T staging is contiguous 16B copies.
__global__ __launch_bounds__(256) void attn_kernel(
    const unsigned short* __restrict__ q,
    const unsigned short* __restrict__ k,
    const unsigned short* __restrict__ vt,
    const int* __restrict__ am,
    unsigned short* __restrict__ out) {
  __shared__ __align__(16) unsigned short Qs[64*64];    // [m][d]
  __shared__ __align__(16) unsigned short VT[64*128];   // [d][j] per slice
  __shared__ __align__(16) unsigned short Pr[64*128];   // [m][j] probs bf16 per slice
  __shared__ float rowsum[64];

  int bid = blockIdx.x;
  int qq    = bid & 3;
  int chunk = (bid >> 2) & 15;
  int hh    = (bid >> 6) % HH;
  int b     = bid / (4*16*HH);

  int tid = threadIdx.x;
  int wave = tid >> 5, lane = tid & 31;
  int ml = lane & 15, hi = lane >> 4;

  int qbase = chunk*CW + qq*64;     // first query seq-pos of this block
  int kwin  = chunk*CW - CW;        // seq-pos of window key j=0 (may be <0)

  // stage Q tile [64x64] (async)
  for (int u = tid; u < 512; u += 256) {
    int row = u >> 3, dq = (u & 7)*8;
    ASYNC_CP16(q + ((size_t)(b*SS + qbase + row))*DD + hh*DH + dq,
               &Qs[row*64 + dq]);
  }
  if (tid < 64) rowsum[tid] = 0.f;
  wait_async0();
  __syncthreads();

  f32x8 oacc[2] = {};   // out tiles: tile = wave*2 + t; mt = tile>>2, nt = tile&3

  for (int sl = 0; sl < 6; ++sl) {    // 6 slices of 128 window keys = 768
    int jbase = sl*128;

    // stage V^T slice [64 d][128 j]: contiguous rows of vt (clamped; masked probs are 0)
    for (int u = tid; u < 1024; u += 256) {
      int d = u >> 4, jc = (u & 15)*8;
      int jg = kwin + jbase + jc;
      jg = jg < 0 ? 0 : (jg > SS-8 ? SS-8 : jg);
      ASYNC_CP16(vt + ((size_t)((b*HH + hh)*DH + d))*SS + jg,
                 &VT[d*128 + jc]);
    }
    wait_async0();
    __syncthreads();

    // scores: wave owns N-tile = wave (16 keys) across 4 M-tiles; K=64 in 2 steps.
    // K rows load directly from global: per-lane contiguous d-range (no transpose).
    f32x8 sacc[4] = {};
    {
      int jn = jbase + wave*16 + ml;
      int kg = kwin + jn;
      int kgc = kg < 0 ? 0 : (kg >= SS ? SS-1 : kg);
      for (int ks = 0; ks < 2; ++ks) {
        FragU bf_;
        const u32x4* kp = (const u32x4*)(k + ((size_t)(b*SS + kgc))*DD
                                         + hh*DH + ks*32 + hi*16);
        bf_.q[0] = kp[0]; bf_.q[1] = kp[1];
        for (int i = 0; i < 4; ++i) {
          FragU af;
          int m = i*16 + ml;
          int ka = ks*32 + hi*16;
          af.q[0] = *(const u32x4*)&Qs[m*64 + ka];
          af.q[1] = *(const u32x4*)&Qs[m*64 + ka + 8];
          sacc[i] = wmma_bf16(af, bf_, sacc[i]);
        }
      }
    }

    // band/pad/attention-mask + exp + rowsum + probs->LDS
    {
      int kg = kwin + jbase + wave*16 + ml;
      bool inb = (kg >= 0) && (kg < SS);
      int mv = inb ? am[(size_t)b*SS + kg] : 0;
      for (int i = 0; i < 4; ++i)
        for (int e = 0; e < 8; ++e) {
          int m = i*16 + e + hi*8;
          int dlt = kg - (qbase + m);
          bool valid = (mv != 0) && (dlt <= CW) && (dlt >= -CW);
          float p = valid ? __expf(sacc[i][e] * 0.125f) : 0.f;  // 1/sqrt(64)
          atomicAdd(&rowsum[m], p);
          Pr[m*128 + wave*16 + ml] = f2bf_bits(p);
        }
    }
    __syncthreads();

    // out += P(64x128) * V(128x64); 16 tiles / 8 waves = 2 each; K=128 in 4 steps
    for (int t = 0; t < 2; ++t) {
      int tile = wave*2 + t, mt = tile >> 2, nt = tile & 3;
      for (int ks = 0; ks < 4; ++ks) {
        FragU af, bfr;
        int m = mt*16 + ml;
        int ka = ks*32 + hi*16;
        af.q[0] = *(const u32x4*)&Pr[m*128 + ka];
        af.q[1] = *(const u32x4*)&Pr[m*128 + ka + 8];
        int n = nt*16 + ml;
        int kb = ks*32 + hi*16;
        bfr.q[0] = *(const u32x4*)&VT[n*128 + kb];
        bfr.q[1] = *(const u32x4*)&VT[n*128 + kb + 8];
        oacc[t] = wmma_bf16(af, bfr, oacc[t]);
      }
    }
    __syncthreads();
  }

  // normalize by row-sum, store bf16 at [b, s, hh*64+d]
  for (int t = 0; t < 2; ++t) {
    int tile = wave*2 + t, mt = tile >> 2, nt = tile & 3;
    int d = nt*16 + ml;
    for (int e = 0; e < 8; ++e) {
      int m = mt*16 + e + hi*8;
      float val = oacc[t][e] / rowsum[m];
      out[((size_t)(b*SS + qbase + m))*DD + hh*DH + d] = f2bf_bits(val);
    }
  }
}

// ---------------- pooler: tanh(h[:,0] @ Wp + bp) ----------------
__global__ void pooler_kernel(const float* __restrict__ h,
                              const float* __restrict__ Wp,
                              const float* __restrict__ bp,
                              float* __restrict__ outp) {
  int n = blockIdx.x*blockDim.x + threadIdx.x;
  if (n >= BB*DD) return;
  int b = n / DD, c = n % DD;
  const float* hr = h + (size_t)b*SS*DD;   // row s=0
  float s = 0.f;
  for (int t = 0; t < DD; ++t) s += hr[t] * Wp[(size_t)t*DD + c];
  outp[n] = tanhf(s + bp[c]);
}

// ---------------- driver ----------------
extern "C" void kernel_launch(void* const* d_in, const int* in_sizes, int n_in,
                              void* d_out, int out_size, void* d_ws, size_t ws_size,
                              hipStream_t stream) {
  (void)in_sizes; (void)n_in; (void)out_size; (void)ws_size;
  const int*   ids  = (const int*)d_in[0];
  const int*   am   = (const int*)d_in[1];
  const float* wemb = (const float*)d_in[2];
  const float* pemb = (const float*)d_in[3];
  const float* temb = (const float*)d_in[4];
  const float* eg   = (const float*)d_in[5];
  const float* ebv  = (const float*)d_in[6];
  const float* Wq   = (const float*)d_in[7];
  const float* bq   = (const float*)d_in[8];
  const float* Wk   = (const float*)d_in[9];
  const float* bk   = (const float*)d_in[10];
  const float* Wv   = (const float*)d_in[11];
  const float* bv   = (const float*)d_in[12];
  const float* Wo   = (const float*)d_in[13];
  const float* bo   = (const float*)d_in[14];
  const float* ln1g = (const float*)d_in[15];
  const float* ln1b = (const float*)d_in[16];
  const float* Wi   = (const float*)d_in[17];
  const float* bi   = (const float*)d_in[18];
  const float* Wf   = (const float*)d_in[19];
  const float* bfw  = (const float*)d_in[20];
  const float* ln2g = (const float*)d_in[21];
  const float* ln2b = (const float*)d_in[22];
  const float* Wp   = (const float*)d_in[23];
  const float* bp   = (const float*)d_in[24];

  // workspace carving (256B aligned)
  char* ws = (char*)d_ws;
  size_t off = 0;
  auto take = [&](size_t bytes) -> void* {
    void* p = ws + off;
    off += (bytes + 255) & ~(size_t)255;
    return p;
  };
  float*          hf  = (float*)         take((size_t)NTOK*DD*4);
  unsigned short* hbf = (unsigned short*)take((size_t)NTOK*DD*2);
  unsigned short* qb  = (unsigned short*)take((size_t)NTOK*DD*2);
  unsigned short* kb  = (unsigned short*)take((size_t)NTOK*DD*2);
  unsigned short* vtb = (unsigned short*)take((size_t)NTOK*DD*2); // V transposed [B,H,DH,S]
  unsigned short* ab  = (unsigned short*)take((size_t)NTOK*DD*2);
  unsigned short* ffb = (unsigned short*)take((size_t)NTOK*FF*2);
  float*          sc0 = (float*)         take((size_t)NTOK*DD*4);
  unsigned short* WqT = (unsigned short*)take((size_t)LL*DD*DD*2);
  unsigned short* WkT = (unsigned short*)take((size_t)LL*DD*DD*2);
  unsigned short* WvT = (unsigned short*)take((size_t)LL*DD*DD*2);
  unsigned short* WoT = (unsigned short*)take((size_t)LL*DD*DD*2);
  unsigned short* WiT = (unsigned short*)take((size_t)LL*DD*FF*2);
  unsigned short* WfT = (unsigned short*)take((size_t)LL*FF*DD*2);

  // per-call weight convert + transpose; bf16 weights (170MB) stay L2-resident
  cvtT_kernel<<<2048, 256, 0, stream>>>(Wq, WqT, DD, DD, (long long)LL*DD*DD);
  cvtT_kernel<<<2048, 256, 0, stream>>>(Wk, WkT, DD, DD, (long long)LL*DD*DD);
  cvtT_kernel<<<2048, 256, 0, stream>>>(Wv, WvT, DD, DD, (long long)LL*DD*DD);
  cvtT_kernel<<<2048, 256, 0, stream>>>(Wo, WoT, DD, DD, (long long)LL*DD*DD);
  cvtT_kernel<<<4096, 256, 0, stream>>>(Wi, WiT, DD, FF, (long long)LL*DD*FF);
  cvtT_kernel<<<4096, 256, 0, stream>>>(Wf, WfT, FF, DD, (long long)LL*FF*DD);

  // embeddings + LN
  embed_kernel<<<NTOK, 256, 0, stream>>>(ids, wemb, pemb, temb, sc0);
  ln_kernel<<<NTOK, 256, 0, stream>>>(sc0, nullptr, nullptr, eg, ebv, hf, hbf);

  dim3 g6(DD/128, NTOK/128);    // 6 x 64
  dim3 g24(FF/128, NTOK/128);   // 24 x 64

  for (int l = 0; l < LL; ++l) {
    // QKV projections (bias fused; V written transposed for attention)
    gemm_bf16_kernel<1><<<g6, 256, 0, stream>>>(hbf, WqT + (size_t)l*DD*DD, bq + l*DD,
                                                nullptr, qb, NTOK, DD, DD);
    gemm_bf16_kernel<1><<<g6, 256, 0, stream>>>(hbf, WkT + (size_t)l*DD*DD, bk + l*DD,
                                                nullptr, kb, NTOK, DD, DD);
    gemm_bf16_kernel<3><<<g6, 256, 0, stream>>>(hbf, WvT + (size_t)l*DD*DD, bv + l*DD,
                                                nullptr, vtb, NTOK, DD, DD);
    // fused sliding-window attention
    attn_kernel<<<BB*HH*16*4, 256, 0, stream>>>(qb, kb, vtb, am, ab);
    // output projection (f32 out), then bias+residual+LN
    gemm_bf16_kernel<0><<<g6, 256, 0, stream>>>(ab, WoT + (size_t)l*DD*DD, nullptr,
                                                sc0, nullptr, NTOK, DD, DD);
    ln_kernel<<<NTOK, 256, 0, stream>>>(sc0, bo + l*DD, hf,
                                        ln1g + l*DD, ln1b + l*DD, hf, hbf);
    // FFN: W_i + bias + exact GELU (bf16), W_f (f32), bias+residual+LN
    gemm_bf16_kernel<2><<<g24, 256, 0, stream>>>(hbf, WiT + (size_t)l*DD*FF, bi + l*FF,
                                                 nullptr, ffb, NTOK, FF, DD);
    gemm_bf16_kernel<0><<<g6, 256, 0, stream>>>(ffb, WfT + (size_t)l*FF*DD, nullptr,
                                                sc0, nullptr, NTOK, DD, FF);
    ln_kernel<<<NTOK, 256, 0, stream>>>(sc0, bfw + l*DD, hf,
                                        ln2g + l*DD, ln2b + l*DD, hf, hbf);
  }

  pooler_kernel<<<(BB*DD + 255)/256, 256, 0, stream>>>(hf, Wp, bp, (float*)d_out);
}
